// EncoderLayer_62354335204021
// MI455X (gfx1250) — matrix operand order repaired and checked by reading, not compiled, not run on hardware
//
#include <hip/hip_runtime.h>

// ---------------------------------------------------------------------------
// Types
// ---------------------------------------------------------------------------
typedef __attribute__((ext_vector_type(16))) __bf16 bf16x16;
typedef __attribute__((ext_vector_type(8)))  __bf16 bf16x8;
typedef __attribute__((ext_vector_type(8)))  float  f32x8;

#define LOG2E 1.44269504088896340736f

static __device__ inline f32x8 wmma_bf16(bf16x16 a, bf16x16 b, f32x8 c) {
  // D = A(16x32 bf16) * B(32x16 bf16) + C(16x16 f32)
  return __builtin_amdgcn_wmma_f32_16x16x32_bf16(false, a, false, b,
                                                 (short)0, c, false, false);
}

// ---------------------------------------------------------------------------
// CDNA5 async copy: global -> LDS (per-lane 16B), tracked by ASYNCcnt.
// LDS address = low 32 bits of generic pointer (flat->LDS truncation, ISA 10.2)
// ---------------------------------------------------------------------------
static __device__ inline void async_load_b128(void* lds, const void* g) {
  unsigned l = (unsigned)(unsigned long long)lds;
  asm volatile("global_load_async_to_lds_b128 %0, %1, off" ::"v"(l), "v"(g)
               : "memory");
}
template <int N>
static __device__ inline void wait_async() {
  asm volatile("s_wait_asynccnt %0" ::"n"(N) : "memory");
}

// ---------------------------------------------------------------------------
// A-fragment (16x32, 16-bit): lane<16 -> row=lane, K = 0..7 & 16..23
//                             lane>=16 -> row=lane-16, K = 8..15 & 24..31
// ---------------------------------------------------------------------------
static __device__ inline bf16x16 load_fragA(const __bf16* base, int row0,
                                            int ldh, int lane) {
  int r  = row0 + (lane & 15);
  int kb = (lane >> 4) << 3;
  const __bf16* p = base + (size_t)r * ldh + kb;
  bf16x8 lo = *(const bf16x8*)p;
  bf16x8 hi = *(const bf16x8*)(p + 16);
  return __builtin_shufflevector(lo, hi, 0, 1, 2, 3, 4, 5, 6, 7, 8, 9, 10, 11,
                                 12, 13, 14, 15);
}

// B-fragment (32x16, 16-bit): lane<16 -> col=lane, K = 0..7 & 16..23
//                             lane>=16 -> col=lane-16, K = 8..15 & 24..31
static __device__ inline bf16x16 load_fragB(const __bf16* base, int col0,
                                            int ldh, int lane) {
  int n  = col0 + (lane & 15);
  int kb = (lane >> 4) << 3;
  const __bf16* p = base + (size_t)n * ldh + kb;
  bf16x8 lo = *(const bf16x8*)p;
  bf16x8 hi = *(const bf16x8*)(p + 16);
  return __builtin_shufflevector(lo, hi, 0, 1, 2, 3, 4, 5, 6, 7, 8, 9, 10, 11,
                                 12, 13, 14, 15);
}

// ---------------------------------------------------------------------------
// Fused single-instruction DPP16 butterfly stages (VOP2 + DPP16):
//   v = max/add(dpp(v), v).  Stages xor1, xor2, ^7, ^15 give a full 16-lane
//   all-reduce (stages 1-2 complete quads; half/row mirror merge them).
// ---------------------------------------------------------------------------
#define DPP_MAX_STAGE(v, ctrl)                              \
  asm("v_max_num_f32_dpp %0, %1, %1 " ctrl                  \
      " row_mask:0xf bank_mask:0xf bound_ctrl:1"            \
      : "=v"(v)                                             \
      : "v"(v))
#define DPP_ADD_STAGE(v, ctrl)                              \
  asm("v_add_f32_dpp %0, %1, %1 " ctrl                      \
      " row_mask:0xf bank_mask:0xf bound_ctrl:1"            \
      : "=v"(v)                                             \
      : "v"(v))

static __device__ inline float red_max16(float v) {
  DPP_MAX_STAGE(v, "quad_perm:[1,0,3,2]");
  DPP_MAX_STAGE(v, "quad_perm:[2,3,0,1]");
  DPP_MAX_STAGE(v, "row_half_mirror");
  DPP_MAX_STAGE(v, "row_mirror");
  return v;
}
static __device__ inline float red_sum16(float v) {
  DPP_ADD_STAGE(v, "quad_perm:[1,0,3,2]");
  DPP_ADD_STAGE(v, "quad_perm:[2,3,0,1]");
  DPP_ADD_STAGE(v, "row_half_mirror");
  DPP_ADD_STAGE(v, "row_mirror");
  return v;
}

// ---------------------------------------------------------------------------
// Elementwise f32 -> bf16
// ---------------------------------------------------------------------------
__global__ __launch_bounds__(256) void cvt_bf16_kernel(
    const float* __restrict__ in, __bf16* __restrict__ out, size_t n) {
  size_t i = (size_t)blockIdx.x * blockDim.x + threadIdx.x;
  if (i < n) out[i] = (__bf16)in[i];
}

// Transpose + convert: w[K][N] f32  ->  wt[N][K] bf16
__global__ __launch_bounds__(256) void transpose_cvt_kernel(
    const float* __restrict__ w, __bf16* __restrict__ wt, int K, int N) {
  __shared__ float t[32][33];
  const int bx = blockIdx.x * 32;  // N
  const int by = blockIdx.y * 32;  // K
  const int tx = threadIdx.x & 31;
  const int ty = threadIdx.x >> 5;  // 0..7
#pragma unroll
  for (int j = 0; j < 4; ++j)
    t[ty + 8 * j][tx] = w[(size_t)(by + ty + 8 * j) * N + bx + tx];
  __syncthreads();
#pragma unroll
  for (int j = 0; j < 4; ++j)
    wt[(size_t)(bx + ty + 8 * j) * K + by + tx] = (__bf16)t[tx][ty + 8 * j];
}

// ---------------------------------------------------------------------------
// WMMA GEMM:  out = A[M,K](bf16) * Bt[N,K](bf16)^T + bias, epilogue variants.
// Double-buffered LDS pipeline fed by global_load_async_to_lds_b128.
// ---------------------------------------------------------------------------
enum { EPI_BF16 = 0, EPI_VT = 1, EPI_RES = 2, EPI_RELU = 3 };

constexpr int BM = 128, BN = 128, BK = 32, LDH = 40;  // padded LDS row (halves)

template <int EPI>
__global__ __launch_bounds__(256) void gemm_bf16_kernel(
    const __bf16* __restrict__ A, const __bf16* __restrict__ Bt,
    const float* __restrict__ bias, const float* __restrict__ res,
    float* __restrict__ outF, __bf16* __restrict__ outB, int M, int N, int K) {
  __shared__ __bf16 As[2][BM * LDH];
  __shared__ __bf16 Bs[2][BN * LDH];
  const int tid  = threadIdx.x;
  const int lane = tid & 31;
  const int wave = tid >> 5;
  const int wm   = wave >> 2;  // 0..1 -> 64 rows each
  const int wn   = wave & 3;   // 0..3 -> 32 cols each
  const int bm   = blockIdx.y * BM;
  const int bn   = blockIdx.x * BN;

  f32x8 acc[4][2];
#pragma unroll
  for (int i = 0; i < 4; ++i)
#pragma unroll
    for (int j = 0; j < 2; ++j) acc[i][j] = f32x8{0, 0, 0, 0, 0, 0, 0, 0};

  // 4 async 16B chunks per thread per tile (2 for A, 2 for B)
  auto issue_tile = [&](int kt, int buf) {
    const int k0 = kt * BK;
#pragma unroll
    for (int i = 0; i < 2; ++i) {
      int c = tid + i * 256;
      int row = c >> 2, col = (c & 3) << 3;
      async_load_b128(&As[buf][row * LDH + col],
                      &A[(size_t)(bm + row) * K + k0 + col]);
      async_load_b128(&Bs[buf][row * LDH + col],
                      &Bt[(size_t)(bn + row) * K + k0 + col]);
    }
  };

  const int NK = K / BK;
  issue_tile(0, 0);
  for (int kt = 0; kt < NK; ++kt) {
    const int cur = kt & 1;
    if (kt + 1 < NK) {
      issue_tile(kt + 1, cur ^ 1);
      wait_async<4>();  // async loads complete in order: tile kt is resident
    } else {
      wait_async<0>();
    }
    __syncthreads();  // tile kt visible to all waves
    bf16x16 bfrag0 = load_fragB(&Bs[cur][0], wn * 32, LDH, lane);
    bf16x16 bfrag1 = load_fragB(&Bs[cur][0], wn * 32 + 16, LDH, lane);
#pragma unroll
    for (int mt = 0; mt < 4; ++mt) {
      bf16x16 afrag = load_fragA(&As[cur][0], wm * 64 + mt * 16, LDH, lane);
      acc[mt][0] = wmma_bf16(afrag, bfrag0, acc[mt][0]);
      acc[mt][1] = wmma_bf16(afrag, bfrag1, acc[mt][1]);
    }
    __syncthreads();  // everyone done reading buf cur before it is re-filled
  }

  // Epilogue. C layout: row = r + 8*(lane>>4), col = lane&15.
  const int half = lane >> 4;
  const int cn   = lane & 15;
#pragma unroll
  for (int mt = 0; mt < 4; ++mt) {
#pragma unroll
    for (int nt = 0; nt < 2; ++nt) {
      int   gn = bn + wn * 32 + nt * 16 + cn;
      float bv = bias[gn];
#pragma unroll
      for (int r = 0; r < 8; ++r) {
        int   gm = bm + wm * 64 + mt * 16 + r + 8 * half;
        float v  = acc[mt][nt][r] + bv;
        if (EPI == EPI_BF16) {
          outB[(size_t)gm * N + gn] = (__bf16)v;
        } else if (EPI == EPI_RELU) {
          outB[(size_t)gm * N + gn] = (__bf16)fmaxf(v, 0.f);
        } else if (EPI == EPI_RES) {
          size_t o = (size_t)gm * N + gn;
          outF[o]  = v + res[o];
        } else {  // EPI_VT: scatter V into [B, H, hd, L] for attention B-frags
          int b = gm >> 11, l = gm & 2047;  // L = 2048
          int h = gn >> 6, d = gn & 63;     // hd = 64
          outB[((size_t)((b * 16 + h) * 64 + d)) * 2048 + l] = (__bf16)v;
        }
      }
    }
  }
}

// ---------------------------------------------------------------------------
// Flash attention, cooperative double-buffered K/V staging via async copies:
//   block = 256 threads = 8 waves, one (b,h), 128 consecutive queries
//   (16 per wave). Loop over 32-key blocks; all 8 waves share LDS-staged
//   K tile (32x64) and V tile (64x32, from pre-transposed V).
// Q,K: [B, L, H*hd] bf16 row-major.  Vt: [B, H, hd, L] bf16.  ctx: [B, L, H*hd]
// ---------------------------------------------------------------------------
constexpr int KLDH = 72;  // K tile row stride in halves (64 + 8 pad)
constexpr int VLDH = 40;  // V tile row stride in halves (32 + 8 pad)

__global__ __launch_bounds__(256) void attn_kernel(
    const __bf16* __restrict__ Q, const __bf16* __restrict__ Km,
    const __bf16* __restrict__ Vt, __bf16* __restrict__ ctx) {
  constexpr int L = 2048, D = 1024, HD = 64, H = 16;
  __shared__ __bf16 Ks[2][32 * KLDH];  // [key][d]
  __shared__ __bf16 Vs[2][64 * VLDH];  // [d][key]
  __shared__ __bf16 Ps[8][16 * LDH];   // per-wave P staging
  const int tid  = threadIdx.x;
  const int lane = tid & 31;
  const int wave = tid >> 5;
  const int bh   = blockIdx.x >> 4;  // 64 (b,h) pairs x 16 query-blocks
  const int h    = bh & (H - 1), b = bh >> 4;
  const int q0   = (blockIdx.x & 15) * 128 + wave * 16;

  const __bf16* qbase = Q + (size_t)b * L * D + h * HD;
  const __bf16* kbase = Km + (size_t)b * L * D + h * HD;
  const __bf16* vbase = Vt + (size_t)(b * H + h) * HD * L;
  __bf16*       cbase = ctx + (size_t)b * L * D + h * HD;

  bf16x16 qf0 = load_fragA(qbase + (size_t)q0 * D, 0, D, lane);       // d 0..31
  bf16x16 qf1 = load_fragA(qbase + (size_t)q0 * D + 32, 0, D, lane);  // d 32..63

  f32x8 o[4];
#pragma unroll
  for (int i = 0; i < 4; ++i) o[i] = f32x8{0, 0, 0, 0, 0, 0, 0, 0};
  float mrow[8], lrow[8];
#pragma unroll
  for (int r = 0; r < 8; ++r) { mrow[r] = -3.0e38f; lrow[r] = 0.f; }

  // K tile: 32 rows x 64 halves = 256 chunks; V tile: 64 x 32 = 256 chunks
  const int krow = tid >> 3, kcol = (tid & 7) << 3;
  const int vrow = tid >> 2, vcol = (tid & 3) << 3;
  auto issue_tile = [&](int kt, int buf) {
    const int kb0 = kt * 32;
    async_load_b128(&Ks[buf][krow * KLDH + kcol],
                    &kbase[(size_t)(kb0 + krow) * D + kcol]);
    async_load_b128(&Vs[buf][vrow * VLDH + vcol],
                    &vbase[(size_t)vrow * L + kb0 + vcol]);
  };

  constexpr int NKT = L / 32;
  issue_tile(0, 0);
  for (int kt = 0; kt < NKT; ++kt) {
    const int cur = kt & 1;
    if (kt + 1 < NKT) {
      issue_tile(kt + 1, cur ^ 1);
      wait_async<2>();
    } else {
      wait_async<0>();
    }
    __syncthreads();

    // ---- scores S = Q K^T * 1/8, 16 queries x 32 keys ----
    f32x8 s[2];
#pragma unroll
    for (int st = 0; st < 2; ++st) {
      bf16x16 kf0 = load_fragB(&Ks[cur][0], st * 16, KLDH, lane);       // 0..31
      bf16x16 kf1 = load_fragB(&Ks[cur][32], st * 16, KLDH, lane);      // 32..63
      f32x8   z   = f32x8{0, 0, 0, 0, 0, 0, 0, 0};
      z           = wmma_bf16(qf0, kf0, z);
      z           = wmma_bf16(qf1, kf1, z);
      s[st]       = z;
    }
    // ---- online softmax; row reductions via fused DPP16 butterflies ----
#pragma unroll
    for (int r = 0; r < 8; ++r) {
      float a0    = s[0][r] * 0.125f, a1 = s[1][r] * 0.125f;
      float mx    = red_max16(fmaxf(a0, a1));
      float mnew  = fmaxf(mrow[r], mx);
      float e0    = exp2f((a0 - mnew) * LOG2E);
      float e1    = exp2f((a1 - mnew) * LOG2E);
      s[0][r]     = e0;
      s[1][r]     = e1;
      float t     = red_sum16(e0 + e1);
      float alpha = exp2f((mrow[r] - mnew) * LOG2E);
      lrow[r]     = lrow[r] * alpha + t;
      mrow[r]     = mnew;
#pragma unroll
      for (int nt = 0; nt < 4; ++nt) o[nt][r] *= alpha;
    }
    // ---- stage P (C layout) -> LDS -> reload as A-fragment ----
    __bf16* ps = &Ps[wave][0];
#pragma unroll
    for (int st = 0; st < 2; ++st)
#pragma unroll
      for (int r = 0; r < 8; ++r)
        ps[(r + 8 * (lane >> 4)) * LDH + st * 16 + (lane & 15)] =
            (__bf16)s[st][r];
    asm volatile("s_wait_dscnt 0" ::: "memory");
    bf16x16 pf = load_fragA(ps, 0, LDH, lane);
    // ---- O += P * V ----
#pragma unroll
    for (int nt = 0; nt < 4; ++nt) {
      bf16x16 vf = load_fragB(&Vs[cur][0], nt * 16, VLDH, lane);
      o[nt]      = wmma_bf16(pf, vf, o[nt]);
    }
    __syncthreads();  // all waves done with buf cur before re-fill
  }
  // ---- normalize and store context ----
  const int half = lane >> 4, cn = lane & 15;
#pragma unroll
  for (int r = 0; r < 8; ++r) {
    float inv = 1.0f / lrow[r];
    int   row = q0 + r + 8 * half;
#pragma unroll
    for (int nt = 0; nt < 4; ++nt)
      cbase[(size_t)row * D + nt * 16 + cn] = (__bf16)(o[nt][r] * inv);
  }
}

// ---------------------------------------------------------------------------
// LayerNorm over D=1024, one 256-thread block per row
// ---------------------------------------------------------------------------
__global__ __launch_bounds__(256) void layernorm_kernel(
    const float* __restrict__ in, const float* __restrict__ g,
    const float* __restrict__ be, float* __restrict__ outF,
    __bf16* __restrict__ outB) {
  constexpr int DD  = 1024;
  const int     row = blockIdx.x;
  const int     tid = threadIdx.x;
  const int lane = tid & 31, wave = tid >> 5;
  const float*     x = in + (size_t)row * DD;
  __shared__ float part[8];
  float            v[4];
  float            s = 0.f;
#pragma unroll
  for (int i = 0; i < 4; ++i) { v[i] = x[tid + 256 * i]; s += v[i]; }
#pragma unroll
  for (int m = 1; m < 32; m <<= 1) s += __shfl_xor(s, m, 32);
  if (!lane) part[wave] = s;
  __syncthreads();
  float mean = 0.f;
#pragma unroll
  for (int i = 0; i < 8; ++i) mean += part[i];
  mean *= (1.f / DD);
  __syncthreads();
  float q = 0.f;
#pragma unroll
  for (int i = 0; i < 4; ++i) { float d = v[i] - mean; q += d * d; }
#pragma unroll
  for (int m = 1; m < 32; m <<= 1) q += __shfl_xor(q, m, 32);
  if (!lane) part[wave] = q;
  __syncthreads();
  float vs = 0.f;
#pragma unroll
  for (int i = 0; i < 8; ++i) vs += part[i];
  float inv = rsqrtf(vs * (1.f / DD) + 1e-5f);
#pragma unroll
  for (int i = 0; i < 4; ++i) {
    int   c = tid + 256 * i;
    float o = (v[i] - mean) * inv * g[c] + be[c];
    if (outF) outF[(size_t)row * DD + c] = o;
    if (outB) outB[(size_t)row * DD + c] = (__bf16)o;
  }
}

// ---------------------------------------------------------------------------
// Host launcher
// ---------------------------------------------------------------------------
extern "C" void kernel_launch(void* const* d_in, const int* in_sizes, int n_in,
                              void* d_out, int out_size, void* d_ws,
                              size_t ws_size, hipStream_t stream) {
  (void)in_sizes; (void)n_in; (void)out_size; (void)ws_size;
  constexpr size_t MT = 8192, D = 1024, F = 4096;  // MT = B*L
  const float* x   = (const float*)d_in[0];
  const float* wq  = (const float*)d_in[1];
  const float* bq  = (const float*)d_in[2];
  const float* wk  = (const float*)d_in[3];
  const float* bk  = (const float*)d_in[4];
  const float* wv  = (const float*)d_in[5];
  const float* bv  = (const float*)d_in[6];
  const float* wo  = (const float*)d_in[7];
  const float* bo  = (const float*)d_in[8];
  const float* w1  = (const float*)d_in[9];
  const float* b1  = (const float*)d_in[10];
  const float* w2  = (const float*)d_in[11];
  const float* b2  = (const float*)d_in[12];
  const float* g1  = (const float*)d_in[13];
  const float* be1 = (const float*)d_in[14];
  const float* g2  = (const float*)d_in[15];
  const float* be2 = (const float*)d_in[16];

  char* p    = (char*)d_ws;
  auto  take = [&p](size_t bytes) {
    char* r = p;
    p += (bytes + 255) & ~(size_t)255;
    return r;
  };
  __bf16* xb   = (__bf16*)take(MT * D * 2);
  __bf16* wqT  = (__bf16*)take(D * D * 2);
  __bf16* wkT  = (__bf16*)take(D * D * 2);
  __bf16* wvT  = (__bf16*)take(D * D * 2);
  __bf16* woT  = (__bf16*)take(D * D * 2);
  __bf16* w1T  = (__bf16*)take(D * F * 2);
  __bf16* w2T  = (__bf16*)take(F * D * 2);
  __bf16* qb   = (__bf16*)take(MT * D * 2);
  __bf16* kb   = (__bf16*)take(MT * D * 2);
  __bf16* vT   = (__bf16*)take(MT * D * 2);
  __bf16* ctx  = (__bf16*)take(MT * D * 2);
  float*  res1 = (float*)take(MT * D * 4);
  float*  hf   = (float*)take(MT * D * 4);
  __bf16* hb   = (__bf16*)take(MT * D * 2);
  __bf16* ff1  = (__bf16*)take(MT * F * 2);
  float*  res2 = (float*)take(MT * D * 4);

  const dim3 blk(256);

  // 1) precision casts / weight transposes
  cvt_bf16_kernel<<<(MT * D) / 256, blk, 0, stream>>>(x, xb, MT * D);
  transpose_cvt_kernel<<<dim3(D / 32, D / 32), blk, 0, stream>>>(wq, wqT, D, D);
  transpose_cvt_kernel<<<dim3(D / 32, D / 32), blk, 0, stream>>>(wk, wkT, D, D);
  transpose_cvt_kernel<<<dim3(D / 32, D / 32), blk, 0, stream>>>(wv, wvT, D, D);
  transpose_cvt_kernel<<<dim3(D / 32, D / 32), blk, 0, stream>>>(wo, woT, D, D);
  transpose_cvt_kernel<<<dim3(F / 32, D / 32), blk, 0, stream>>>(w1, w1T, D, F);
  transpose_cvt_kernel<<<dim3(D / 32, F / 32), blk, 0, stream>>>(w2, w2T, F, D);

  // 2) Q/K/V projections (V stored transposed per head for attention)
  gemm_bf16_kernel<EPI_BF16><<<dim3(D / BN, MT / BM), blk, 0, stream>>>(
      xb, wqT, bq, nullptr, nullptr, qb, (int)MT, (int)D, (int)D);
  gemm_bf16_kernel<EPI_BF16><<<dim3(D / BN, MT / BM), blk, 0, stream>>>(
      xb, wkT, bk, nullptr, nullptr, kb, (int)MT, (int)D, (int)D);
  gemm_bf16_kernel<EPI_VT><<<dim3(D / BN, MT / BM), blk, 0, stream>>>(
      xb, wvT, bv, nullptr, nullptr, vT, (int)MT, (int)D, (int)D);

  // 3) flash attention -> context (bf16)
  attn_kernel<<<1024, blk, 0, stream>>>(qb, kb, vT, ctx);

  // 4) output projection + residual (f32)
  gemm_bf16_kernel<EPI_RES><<<dim3(D / BN, MT / BM), blk, 0, stream>>>(
      ctx, woT, bo, x, res1, nullptr, (int)MT, (int)D, (int)D);

  // 5) LN1 -> h (f32 for residual, bf16 for FFN GEMM)
  layernorm_kernel<<<MT, blk, 0, stream>>>(res1, g1, be1, hf, hb);

  // 6) FFN
  gemm_bf16_kernel<EPI_RELU><<<dim3(F / BN, MT / BM), blk, 0, stream>>>(
      hb, w1T, b1, nullptr, nullptr, ff1, (int)MT, (int)F, (int)D);
  gemm_bf16_kernel<EPI_RES><<<dim3(D / BN, MT / BM), blk, 0, stream>>>(
      ff1, w2T, b2, hf, res2, nullptr, (int)MT, (int)D, (int)F);

  // 7) LN2 -> output (f32)
  layernorm_kernel<<<MT, blk, 0, stream>>>(res2, g2, be2, (float*)d_out,
                                           nullptr);
}
